// GNN_19301583028903
// MI455X (gfx1250) — compile-verified, matrix operand order checked by reading, not run on hardware
//
#include <hip/hip_runtime.h>
#include <hip/hip_bf16.h>

// ---------------------------------------------------------------------------
// GCN forward on MI455X (gfx1250, wave32).
//   layer: h = X@W ; agg[dst] += h[src]*dinv[src]*dinv[dst] ; += h*dinv^2 + b
//   GEMMs via v_wmma_f32_16x16x32_bf16; aggregation via fp32 global atomics
//   (agg arrays are 51.2MB -> L2 resident on the 192MB L2).
// ---------------------------------------------------------------------------

#define NN 50000
#define NE 800000
#define DD 256
#define NC 10
#define NG 128

typedef __attribute__((ext_vector_type(16))) __bf16 v16bf;
typedef __attribute__((ext_vector_type(8)))  float  v8f;

// ---------------- conversion kernels ----------------

// fp32 -> bf16, 4 elements / thread
__global__ __launch_bounds__(256) void cvt_bf16_kernel(
    const float* __restrict__ src, __bf16* __restrict__ dst, int n4) {
  int t = blockIdx.x * 256 + threadIdx.x;
  if (t >= n4) return;
  float4 v = ((const float4*)src)[t];
  union { __bf16 h[4]; uint2 u; } o;
  o.h[0] = (__bf16)v.x; o.h[1] = (__bf16)v.y;
  o.h[2] = (__bf16)v.z; o.h[3] = (__bf16)v.w;
  ((uint2*)dst)[t] = o.u;
}

// W[k][n] fp32 -> Wt[n][k] bf16   (256x256)
__global__ __launch_bounds__(256) void transpose_cvt_kernel(
    const float* __restrict__ W, __bf16* __restrict__ Wt) {
  int t = blockIdx.x * 256 + threadIdx.x;      // t < 65536
  int n = t & 255, k = t >> 8;
  Wt[n * DD + k] = (__bf16)W[k * DD + n];
}

// relu(agg) -> bf16 (input for next layer), 4 elements / thread
__global__ __launch_bounds__(256) void relu_cvt_kernel(
    const float* __restrict__ src, __bf16* __restrict__ dst, int n4) {
  int t = blockIdx.x * 256 + threadIdx.x;
  if (t >= n4) return;
  float4 v = ((const float4*)src)[t];
  union { __bf16 h[4]; uint2 u; } o;
  o.h[0] = (__bf16)fmaxf(v.x, 0.f); o.h[1] = (__bf16)fmaxf(v.y, 0.f);
  o.h[2] = (__bf16)fmaxf(v.z, 0.f); o.h[3] = (__bf16)fmaxf(v.w, 0.f);
  ((uint2*)dst)[t] = o.u;
}

// ---------------- degree / normalization ----------------

__global__ __launch_bounds__(256) void deg_init_kernel(float* __restrict__ deg) {
  int i = blockIdx.x * 256 + threadIdx.x;
  if (i < NN) deg[i] = 1.0f;                   // self-loop
}

__global__ __launch_bounds__(256) void deg_acc_kernel(
    const int* __restrict__ dst, float* __restrict__ deg) {
  int e = blockIdx.x * 256 + threadIdx.x;
  if (e < NE) atomicAdd(&deg[dst[e]], 1.0f);
}

__global__ __launch_bounds__(256) void dinv_kernel(float* __restrict__ deg) {
  int i = blockIdx.x * 256 + threadIdx.x;
  if (i < NN) deg[i] = rsqrtf(deg[i]);         // in-place -> dinv
}

// ---------------- WMMA GEMM + fused self-loop/bias epilogue ----------------
// A : [M,256] bf16 row-major (node features)
// Bt: [256,256] bf16, Bt[n][k] = W[k][n]
// H  <- A@W (fp32), AGG <- H*dinv[m]^2 + bias[n]
// One 16x16 output tile per wave; 8 waves/block; M=50000=3125*16 tiles of 16.
__global__ __launch_bounds__(256) void gemm_wmma_kernel(
    const __bf16* __restrict__ A, const __bf16* __restrict__ Bt,
    const float* __restrict__ bias, const float* __restrict__ dinv,
    float* __restrict__ H, float* __restrict__ AGG) {
  const int lane = threadIdx.x & 31;
  const int wid  = threadIdx.x >> 5;
  const int tile = blockIdx.x * 8 + wid;       // 3125*16 = 50000 tiles
  const int mT   = tile >> 4;
  const int nT   = tile & 15;
  const int half = lane >> 4;
  const int lrow = lane & 15;
  const int m0 = mT * 16, n0 = nT * 16;

  const __bf16* arow = A  + (size_t)(m0 + lrow) * DD;
  const __bf16* brow = Bt + (size_t)(n0 + lrow) * DD;

  v8f c = {};
#pragma unroll
  for (int ks = 0; ks < DD; ks += 32) {
    // A frag: lane holds row m0+lrow; element i <-> k = ks + 8*half + (i<8?i:i+8)
    // B frag: lane holds col n0+lrow; same k mapping (loaded from transposed W)
    const int kb0 = ks + 8 * half;
    const int kb1 = kb0 + 16;
    union { v16bf v; uint4 u[2]; } a, b;
    a.u[0] = *(const uint4*)(arow + kb0);
    a.u[1] = *(const uint4*)(arow + kb1);
    b.u[0] = *(const uint4*)(brow + kb0);
    b.u[1] = *(const uint4*)(brow + kb1);
    c = __builtin_amdgcn_wmma_f32_16x16x32_bf16(
            /*neg_a=*/false, a.v, /*neg_b=*/false, b.v,
            /*c_mod=*/(short)0, c, /*reuse_a=*/false, /*reuse_b=*/false);
  }

  // C/D layout: VGPR r, lane -> m = m0 + 8*half + r, n = n0 + lrow
  const float bn = bias[n0 + lrow];
#pragma unroll
  for (int r = 0; r < 8; ++r) {
    const int m = m0 + 8 * half + r;
    const size_t idx = (size_t)m * DD + (n0 + lrow);
    const float di = dinv[m];
    H[idx]   = c[r];
    AGG[idx] = c[r] * (di * di) + bn;          // self-loop + bias fused
  }
}

// ---------------- edge scatter: agg[dst] += h[src]*dinv[src]*dinv[dst] -----
// 64 lanes per edge, float4 per lane, fp32 atomics into L2-resident agg.
__global__ __launch_bounds__(256) void edge_scatter_kernel(
    const int* __restrict__ src, const int* __restrict__ dst,
    const float* __restrict__ dinv, const float* __restrict__ H,
    float* __restrict__ AGG) {
  int gid = blockIdx.x * 256 + threadIdx.x;
  if (gid >= NE * 64) return;
  const int e = gid >> 6;
  const int q = gid & 63;
  const int s = src[e], d = dst[e];
  const float coef = dinv[s] * dinv[d];
  const float4 v = ((const float4*)H)[(size_t)s * 64 + q];
  float* base = AGG + (size_t)d * DD + q * 4;
  atomicAdd(base + 0, v.x * coef);
  atomicAdd(base + 1, v.y * coef);
  atomicAdd(base + 2, v.z * coef);
  atomicAdd(base + 3, v.w * coef);
}

// ---------------- global mean pool + classifier ----------------

__global__ __launch_bounds__(256) void pool_zero_kernel(
    float* __restrict__ pool, float* __restrict__ cnt) {
  int i = blockIdx.x * 256 + threadIdx.x;
  if (i < NG * DD) pool[i] = 0.0f;
  if (i < NG) cnt[i] = 0.0f;
}

__global__ __launch_bounds__(256) void pool_acc_kernel(
    const int* __restrict__ batch, const float* __restrict__ AGG,
    float* __restrict__ pool, float* __restrict__ cnt) {
  int gid = blockIdx.x * 256 + threadIdx.x;
  if (gid >= NN * 64) return;
  const int i = gid >> 6;
  const int q = gid & 63;
  const int g = batch[i];
  const float4 v = ((const float4*)AGG)[(size_t)i * 64 + q];
  float* base = pool + (size_t)g * DD + q * 4;
  atomicAdd(base + 0, v.x);
  atomicAdd(base + 1, v.y);
  atomicAdd(base + 2, v.z);
  atomicAdd(base + 3, v.w);
  if (q == 0) atomicAdd(&cnt[g], 1.0f);
}

__global__ __launch_bounds__(256) void classifier_kernel(
    const float* __restrict__ pool, const float* __restrict__ cnt,
    const float* __restrict__ Wl, const float* __restrict__ bl,
    float* __restrict__ out) {
  int t = blockIdx.x * 256 + threadIdx.x;
  if (t >= NG * NC) return;
  const int g = t / NC, c = t % NC;
  const float inv = 1.0f / fmaxf(cnt[g], 1.0f);
  float acc = 0.0f;
  for (int k = 0; k < DD; ++k) acc += pool[g * DD + k] * Wl[k * NC + c];
  out[g * NC + c] = acc * inv + bl[c];
}

// ---------------------------------------------------------------------------

extern "C" void kernel_launch(void* const* d_in, const int* in_sizes, int n_in,
                              void* d_out, int out_size, void* d_ws, size_t ws_size,
                              hipStream_t stream) {
  (void)in_sizes; (void)n_in; (void)out_size; (void)ws_size;

  const float* x   = (const float*)d_in[0];
  const int*   ei  = (const int*)d_in[1];     // [2, NE]: row0 = src, row1 = dst
  const int*   bat = (const int*)d_in[2];
  const float* W1  = (const float*)d_in[3];
  const float* b1  = (const float*)d_in[4];
  const float* W2  = (const float*)d_in[5];
  const float* b2  = (const float*)d_in[6];
  const float* Wl  = (const float*)d_in[7];
  const float* bl  = (const float*)d_in[8];
  float* out = (float*)d_out;

  const int* src = ei;
  const int* dst = ei + NE;

  // ---- workspace layout (256B-aligned offsets), ~128.6 MB total ----
  char* ws = (char*)d_ws;
  __bf16* xb   = (__bf16*)(ws + 0);                  // 25,600,000 B (layer input, bf16)
  float*  H    = (float*) (ws + 25600000);           // 51,200,000 B
  float*  AGG  = (float*) (ws + 76800000);           // 51,200,000 B
  __bf16* W1t  = (__bf16*)(ws + 128000000);          //    131,072 B
  __bf16* W2t  = (__bf16*)(ws + 128131072);          //    131,072 B
  float*  dinv = (float*) (ws + 128262144);          //    200,192 B (deg -> dinv)
  float*  pool = (float*) (ws + 128462336);          //    131,072 B
  float*  cnt  = (float*) (ws + 128593408);          //        512 B

  const int nb_feat4 = (NN * DD / 4 + 255) / 256;    // 12500
  const int nb_node  = (NN + 255) / 256;             // 196
  const int nb_edge  = (NE + 255) / 256;             // 3125
  const int nb_gemm  = (NN / 16) * 16 / 8;           // 6250 (8 waves/block, 1 tile/wave)
  const int nb_scat  = NE * 64 / 256;                // 200000
  const int nb_pool  = NN * 64 / 256;                // 12500

  // precision prep
  cvt_bf16_kernel<<<nb_feat4, 256, 0, stream>>>(x, xb, NN * DD / 4);
  transpose_cvt_kernel<<<256, 256, 0, stream>>>(W1, W1t);
  transpose_cvt_kernel<<<256, 256, 0, stream>>>(W2, W2t);

  // degree normalization
  deg_init_kernel<<<nb_node, 256, 0, stream>>>(dinv);
  deg_acc_kernel<<<nb_edge, 256, 0, stream>>>(dst, dinv);
  dinv_kernel<<<nb_node, 256, 0, stream>>>(dinv);

  // ---- layer 1 ----
  gemm_wmma_kernel<<<nb_gemm, 256, 0, stream>>>(xb, W1t, b1, dinv, H, AGG);
  edge_scatter_kernel<<<nb_scat, 256, 0, stream>>>(src, dst, dinv, H, AGG);
  relu_cvt_kernel<<<nb_feat4, 256, 0, stream>>>(AGG, xb, NN * DD / 4);

  // ---- layer 2 ----
  gemm_wmma_kernel<<<nb_gemm, 256, 0, stream>>>(xb, W2t, b2, dinv, H, AGG);
  edge_scatter_kernel<<<nb_scat, 256, 0, stream>>>(src, dst, dinv, H, AGG);

  // ---- pool + classifier ----
  pool_zero_kernel<<<(NG * DD + 255) / 256, 256, 0, stream>>>(pool, cnt);
  pool_acc_kernel<<<nb_pool, 256, 0, stream>>>(bat, AGG, pool, cnt);
  classifier_kernel<<<(NG * NC + 255) / 256, 256, 0, stream>>>(pool, cnt, Wl, bl, out);
}
